// MultiHeadAlignedTokenAttention_1125281432102
// MI455X (gfx1250) — compile-verified, matrix operand order checked by reading, not run on hardware
//
#include <hip/hip_runtime.h>
#include <hip/hip_bf16.h>

// MultiHeadAlignedTokenAttention for MI455X (gfx1250, wave32, WMMA).
// All GEMMs use v_wmma_f32_16x16x32_f16 (f16 inputs, f32 accumulate).
// Pipeline (all f16 intermediates in workspace):
//   1) prep:   Wq/Wk/Wv/Wo fp32 [in][out] -> f16 Wt [out][in]
//   2) proj:   Qh = ini_q@Wq+bq   -> [b,h,q,d]      (16-row blocks, 8 waves x 6 col-tiles)
//              Kh,Vt = ini_k@{Wk,Wv}+{bk,bv} fused (ini_k read ONCE, DUAL templated
//              so the second WMMA stream is branch-free)
//              Kh -> [b,h,tok,d]  (B^T layout for Q.K^T)
//              Vt -> [b,h,s,d,t]  (B^T layout for alpha.V)
//   3) attn:   per (b,h,s): scores in WMMA C-frags, scale+mask, softmax over t
//              via shfl_xor half-wave reductions, alpha->LDS->A-frags, alpha@V
//   4) oproj:  res@Wo+bo fused with LayerNorm (cross-wave LDS reduction), fp32 out.
// A-tile staging is ping-pong double-buffered (1 barrier per K-step); weight
// stream prefetched at WGP scope (locality 3 -> global_prefetch_b8, all levels).

typedef __attribute__((ext_vector_type(16))) _Float16 v16h;
typedef __attribute__((ext_vector_type(8)))  _Float16 v8h;
typedef __attribute__((ext_vector_type(8)))  float    v8f;

#define HIDDEN 768
#define NHEAD  12
#define HDIM   64
#define BB     4
#define QQ     64
#define SS     64
#define TT     128

union FragH { v16h v; v8h h[2]; };

// 16x32 f16 fragment from row-major storage (ld in elements).
// lane: row m = lane&15, K halves {kb..kb+7} u {16+kb..16+kb+7}, kb=(lane>>4)*8
__device__ __forceinline__ v16h load_frag(const _Float16* __restrict__ base, int ld,
                                          int m, int kb) {
  FragH u;
  const _Float16* p = base + (long)m * ld + kb;
  u.h[0] = *(const v8h*)(p);
  u.h[1] = *(const v8h*)(p + 16);
  return u.v;
}
__device__ __forceinline__ v16h load_frag_lds(const _Float16* base, int ld, int m, int kb) {
  FragH u;
  const _Float16* p = base + m * ld + kb;
  u.h[0] = *(const v8h*)(p);
  u.h[1] = *(const v8h*)(p + 16);
  return u.v;
}

// ---------------- 1) weight prep: fp32 [in][out] -> f16 [out][in] ----------------
__global__ void prep_w_kernel(const float* __restrict__ W0, const float* __restrict__ W1,
                              const float* __restrict__ W2, const float* __restrict__ W3,
                              _Float16* __restrict__ T0, _Float16* __restrict__ T1,
                              _Float16* __restrict__ T2, _Float16* __restrict__ T3) {
  const float* src = (blockIdx.y == 0) ? W0 : (blockIdx.y == 1) ? W1 : (blockIdx.y == 2) ? W2 : W3;
  _Float16*    dst = (blockIdx.y == 0) ? T0 : (blockIdx.y == 1) ? T1 : (blockIdx.y == 2) ? T2 : T3;
  long i = (long)blockIdx.x * blockDim.x + threadIdx.x;
  if (i >= (long)HIDDEN * HIDDEN) return;
  int o = (int)(i / HIDDEN);
  int k = (int)(i % HIDDEN);
  dst[i] = (_Float16)src[(long)k * HIDDEN + o];   // transpose
}

// ---------------- 2) projection(s): Y = X @ W + b, WMMA, layout-scatter epilogue ----
template <int MODE>
__device__ __forceinline__ void store_proj(_Float16* __restrict__ out,
                                           int r, int h, int d, float y) {
  if (MODE == 0) {                      // Q: rows = b*64+q -> [b,h,q,d]
    int b = r >> 6, q = r & 63;
    out[(((long)b * NHEAD + h) * QQ + q) * HDIM + d] = (_Float16)y;
  } else if (MODE == 1) {               // K: rows = b*8192+tok -> [b,h,tok,d]
    int b = r >> 13, tok = r & 8191;
    out[(((long)b * NHEAD + h) * (SS * TT) + tok) * HDIM + d] = (_Float16)y;
  } else {                              // V: rows = b*8192+tok -> [b,h,s,d,t]
    int b = r >> 13, tok = r & 8191;
    int s = tok >> 7, t = tok & 127;
    out[((((long)b * NHEAD + h) * SS + s) * HDIM + d) * TT + t] = (_Float16)y;
  }
}

__device__ __forceinline__ void stage_tile(_Float16* dst, const float* __restrict__ X,
                                           long row0, int k0, int tid) {
  // 16 rows x 32 k, fp32 -> f16, 256 threads (2 elems each)
  for (int i = tid; i < 512; i += 256) {
    int r = i >> 5, kk = i & 31;
    dst[r * 32 + kk] = (_Float16)X[(row0 + r) * HIDDEN + k0 + kk];
  }
}

template <bool DUAL, int MODE1>
__global__ __launch_bounds__(256) void proj_kernel(
    const float* __restrict__ X,
    const _Float16* __restrict__ Wt1, const float* __restrict__ bias1,
    _Float16* __restrict__ out1,
    const _Float16* __restrict__ Wt2, const float* __restrict__ bias2,
    _Float16* __restrict__ out2) {
  __shared__ _Float16 lds_a[2][16 * 32];      // ping-pong A tile (2 KB)
  const int tid  = threadIdx.x;
  const int wave = tid >> 5;
  const int lane = tid & 31;
  const int m    = lane & 15;
  const int kb   = (lane >> 4) * 8;
  const long row0 = (long)blockIdx.x * 16;
  const int col0 = wave * 96;                 // 6 tiles of 16 per wave

  v8f c1[6], c2[6];
  v8f zero8 = {0.f, 0.f, 0.f, 0.f, 0.f, 0.f, 0.f, 0.f};
  for (int i = 0; i < 6; ++i) { c1[i] = zero8; c2[i] = zero8; }

  stage_tile(lds_a[0], X, row0, 0, tid);
  int cur = 0;
  for (int k0 = 0; k0 < HIDDEN; k0 += 32) {
    __syncthreads();                          // stage of lds_a[cur] complete
    if (k0 + 32 < HIDDEN)                     // overlap next stage with WMMAs
      stage_tile(lds_a[cur ^ 1], X, row0, k0 + 32, tid);
    v16h a = load_frag_lds(lds_a[cur], 32, m, kb);
    for (int tI = 0; tI < 6; ++tI) {
      int n = col0 + tI * 16 + m;
      const _Float16* bp1 = Wt1 + (long)n * HIDDEN + k0 + kb;
      __builtin_prefetch(bp1 + 32, 0, 3);     // WGP-scope prefetch of next K-step
      FragH b1;
      b1.h[0] = *(const v8h*)(bp1);
      b1.h[1] = *(const v8h*)(bp1 + 16);
      c1[tI] = __builtin_amdgcn_wmma_f32_16x16x32_f16(false, a, false, b1.v,
                                                      (short)0, c1[tI], false, false);
      if (DUAL) {
        const _Float16* bp2 = Wt2 + (long)n * HIDDEN + k0 + kb;
        __builtin_prefetch(bp2 + 32, 0, 3);
        FragH b2;
        b2.h[0] = *(const v8h*)(bp2);
        b2.h[1] = *(const v8h*)(bp2 + 16);
        c2[tI] = __builtin_amdgcn_wmma_f32_16x16x32_f16(false, a, false, b2.v,
                                                        (short)0, c2[tI], false, false);
      }
    }
    cur ^= 1;
  }
  // epilogue: bias + layout scatter (C-frag: col = lane&15, rows = (lane>>4)*8+v)
  const int rb = (lane >> 4) * 8;
  for (int tI = 0; tI < 6; ++tI) {
    int col = col0 + tI * 16 + m;
    float b1v = bias1[col];
    float b2v = DUAL ? bias2[col] : 0.f;
    int h = col >> 6, d = col & 63;
    for (int v = 0; v < 8; ++v) {
      int r = (int)row0 + rb + v;
      store_proj<MODE1>(out1, r, h, d, c1[tI][v] + b1v);
      if (DUAL) store_proj<2>(out2, r, h, d, c2[tI][v] + b2v);
    }
  }
}

// ---------------- 3) attention: scores + mask + softmax + alpha@V ----------------
__global__ __launch_bounds__(128) void attn_kernel(
    const _Float16* __restrict__ Qh, const _Float16* __restrict__ Kh,
    const _Float16* __restrict__ Vt, const float* __restrict__ mask,
    _Float16* __restrict__ res) {
  __shared__ _Float16 lds_alpha[QQ * TT];     // 64 x 128 f16 = 16 KB
  const int tid  = threadIdx.x;
  const int wave = tid >> 5;                  // 4 waves, 16 q-rows each
  const int lane = tid & 31;
  const int m    = lane & 15;
  const int kb   = (lane >> 4) * 8;
  const int rb   = (lane >> 4) * 8;

  const int bid = blockIdx.x;                 // ((b*12+h)*64 + s)
  const int s   = bid & 63;
  const int bh  = bid >> 6;
  const int b   = bh / NHEAD;
  const int h   = bh % NHEAD;

  const _Float16* qbase = Qh + (long)bh * QQ * HDIM;                         // [q][d]
  const _Float16* kbase = Kh + ((long)bh * (SS * TT) + (long)s * TT) * HDIM; // [t][d]
  const _Float16* vbase = Vt + ((long)bh * SS + s) * (HDIM * TT);            // [d][t]
  const float*    mrow  = mask + ((long)b * SS + s) * TT;

  // scores: 8 t-tiles of 16, K over d=64 in 2 steps
  v8f sc[8];
  v8f zero8 = {0.f, 0.f, 0.f, 0.f, 0.f, 0.f, 0.f, 0.f};
  for (int i = 0; i < 8; ++i) sc[i] = zero8;
  for (int k0 = 0; k0 < HDIM; k0 += 32) {
    v16h a = load_frag(qbase + (long)wave * 16 * HDIM, HDIM, m, k0 + kb);
    for (int tt = 0; tt < 8; ++tt) {
      v16h bf = load_frag(kbase + (long)tt * 16 * HDIM, HDIM, m, k0 + kb);
      sc[tt] = __builtin_amdgcn_wmma_f32_16x16x32_f16(false, a, false, bf,
                                                      (short)0, sc[tt], false, false);
    }
  }
  // scale + mask bias; rowwise max over t=128 (16-lane half-wave reduction)
  float rmax[8];
  for (int v = 0; v < 8; ++v) rmax[v] = -3.4e38f;
  for (int tt = 0; tt < 8; ++tt) {
    float mv = mrow[tt * 16 + m] * -10000.0f;
    for (int v = 0; v < 8; ++v) {
      float x = sc[tt][v] * 0.125f + mv;      // SCALE = 1/sqrt(64)
      sc[tt][v] = x;
      rmax[v] = fmaxf(rmax[v], x);
    }
  }
  for (int off = 1; off < 16; off <<= 1)
    for (int v = 0; v < 8; ++v) rmax[v] = fmaxf(rmax[v], __shfl_xor(rmax[v], off, 32));
  float rsum[8];
  for (int v = 0; v < 8; ++v) rsum[v] = 0.f;
  for (int tt = 0; tt < 8; ++tt)
    for (int v = 0; v < 8; ++v) {
      float e = __expf(sc[tt][v] - rmax[v]);
      sc[tt][v] = e;
      rsum[v] += e;
    }
  for (int off = 1; off < 16; off <<= 1)
    for (int v = 0; v < 8; ++v) rsum[v] += __shfl_xor(rsum[v], off, 32);
  // alpha -> LDS [q][t] (C-frag scatter), then reload as A-frags
  for (int tt = 0; tt < 8; ++tt)
    for (int v = 0; v < 8; ++v) {
      int q = wave * 16 + rb + v;
      lds_alpha[q * TT + tt * 16 + m] = (_Float16)(sc[tt][v] / rsum[v]);
    }
  __syncthreads();
  // out[16q x 64d] = alpha[16q x 128t] @ V^T-layout, K over t in 4 steps
  v8f o[4];
  for (int i = 0; i < 4; ++i) o[i] = zero8;
  for (int kt = 0; kt < 4; ++kt) {
    v16h a = load_frag_lds(lds_alpha + wave * 16 * TT, TT, m, kt * 32 + kb);
    for (int dd = 0; dd < 4; ++dd) {
      v16h bf = load_frag(vbase + (long)dd * 16 * TT, TT, m, kt * 32 + kb);
      o[dd] = __builtin_amdgcn_wmma_f32_16x16x32_f16(false, a, false, bf,
                                                     (short)0, o[dd], false, false);
    }
  }
  // res[b,q,s,h*64+d] f16
  for (int dd = 0; dd < 4; ++dd) {
    int col = h * HDIM + dd * 16 + m;
    for (int v = 0; v < 8; ++v) {
      int q = wave * 16 + rb + v;
      res[(((long)b * QQ + q) * SS + s) * HIDDEN + col] = (_Float16)o[dd][v];
    }
  }
}

// ---------------- 4) O-projection + LayerNorm (fused) ----------------
__global__ __launch_bounds__(256) void oproj_ln_kernel(
    const _Float16* __restrict__ Res, const _Float16* __restrict__ Wot,
    const float* __restrict__ bo, const float* __restrict__ gamma,
    const float* __restrict__ beta, float* __restrict__ out) {
  __shared__ float s_sum[16], s_sq[16], s_mean[16], s_rstd[16];
  const int tid  = threadIdx.x;
  const int wave = tid >> 5;
  const int lane = tid & 31;
  const int m    = lane & 15;
  const int kb   = (lane >> 4) * 8;
  const int rb   = (lane >> 4) * 8;
  const long row0 = (long)blockIdx.x * 16;
  const int col0  = wave * 96;

  if (tid < 16) { s_sum[tid] = 0.f; s_sq[tid] = 0.f; }
  __syncthreads();

  v8f c[6];
  v8f zero8 = {0.f, 0.f, 0.f, 0.f, 0.f, 0.f, 0.f, 0.f};
  for (int i = 0; i < 6; ++i) c[i] = zero8;
  for (int k0 = 0; k0 < HIDDEN; k0 += 32) {
    v16h a = load_frag(Res + row0 * HIDDEN, HIDDEN, m, k0 + kb);
    for (int tI = 0; tI < 6; ++tI) {
      const _Float16* bp = Wot + (long)(col0 + tI * 16 + m) * HIDDEN + k0 + kb;
      __builtin_prefetch(bp + 32, 0, 3);      // WGP-scope prefetch of next K-step
      FragH bf;
      bf.h[0] = *(const v8h*)(bp);
      bf.h[1] = *(const v8h*)(bp + 16);
      c[tI] = __builtin_amdgcn_wmma_f32_16x16x32_f16(false, a, false, bf.v,
                                                     (short)0, c[tI], false, false);
    }
  }
  // bias add + per-row partial sums (this lane: 6 cols x 8 rows)
  float psum[8], psq[8];
  for (int v = 0; v < 8; ++v) { psum[v] = 0.f; psq[v] = 0.f; }
  for (int tI = 0; tI < 6; ++tI) {
    float bv = bo[col0 + tI * 16 + m];
    for (int v = 0; v < 8; ++v) {
      float y = c[tI][v] + bv;
      c[tI][v] = y;
      psum[v] += y;
      psq[v]  += y * y;
    }
  }
  for (int off = 1; off < 16; off <<= 1)
    for (int v = 0; v < 8; ++v) {
      psum[v] += __shfl_xor(psum[v], off, 32);
      psq[v]  += __shfl_xor(psq[v],  off, 32);
    }
  if (m == 0)
    for (int v = 0; v < 8; ++v) {
      atomicAdd(&s_sum[rb + v], psum[v]);
      atomicAdd(&s_sq [rb + v], psq[v]);
    }
  __syncthreads();
  if (tid < 16) {
    float mu  = s_sum[tid] * (1.0f / HIDDEN);
    float var = s_sq[tid] * (1.0f / HIDDEN) - mu * mu;
    s_mean[tid] = mu;
    s_rstd[tid] = rsqrtf(var + 1e-12f);
  }
  __syncthreads();
  for (int tI = 0; tI < 6; ++tI) {
    int col = col0 + tI * 16 + m;
    float g = gamma[col], bt = beta[col];
    for (int v = 0; v < 8; ++v) {
      int rloc = rb + v;
      out[(row0 + rloc) * HIDDEN + col] = (c[tI][v] - s_mean[rloc]) * s_rstd[rloc] * g + bt;
    }
  }
}

// ---------------- host launcher ----------------
extern "C" void kernel_launch(void* const* d_in, const int* in_sizes, int n_in,
                              void* d_out, int out_size, void* d_ws, size_t ws_size,
                              hipStream_t stream) {
  const float* ini_q = (const float*)d_in[0];
  const float* ini_k = (const float*)d_in[1];
  const float* mask  = (const float*)d_in[2];
  const float* Wq    = (const float*)d_in[3];
  const float* bq    = (const float*)d_in[4];
  const float* Wk    = (const float*)d_in[5];
  const float* bk    = (const float*)d_in[6];
  const float* Wv    = (const float*)d_in[7];
  const float* bv    = (const float*)d_in[8];
  const float* Wo    = (const float*)d_in[9];
  const float* bo    = (const float*)d_in[10];
  const float* gamma = (const float*)d_in[11];
  const float* beta  = (const float*)d_in[12];

  char* ws = (char*)d_ws;
  size_t off = 0;
  auto alloc = [&](size_t bytes) -> char* {
    char* p = ws + off;
    off = (off + bytes + 255) & ~(size_t)255;
    return p;
  };
  const size_t WBYTES = (size_t)HIDDEN * HIDDEN * sizeof(_Float16);
  _Float16* Wtq = (_Float16*)alloc(WBYTES);
  _Float16* Wtk = (_Float16*)alloc(WBYTES);
  _Float16* Wtv = (_Float16*)alloc(WBYTES);
  _Float16* Wto = (_Float16*)alloc(WBYTES);
  _Float16* Qh  = (_Float16*)alloc((size_t)BB * NHEAD * QQ * HDIM * sizeof(_Float16));
  _Float16* Kh  = (_Float16*)alloc((size_t)BB * NHEAD * SS * TT * HDIM * sizeof(_Float16));
  _Float16* Vt  = (_Float16*)alloc((size_t)BB * NHEAD * SS * HDIM * TT * sizeof(_Float16));
  _Float16* Res = (_Float16*)alloc((size_t)BB * QQ * SS * HIDDEN * sizeof(_Float16));

  // 1) weight transpose/convert (4 matrices)
  {
    dim3 grid((HIDDEN * HIDDEN + 255) / 256, 4);
    prep_w_kernel<<<grid, 256, 0, stream>>>(Wq, Wk, Wv, Wo, Wtq, Wtk, Wtv, Wto);
  }
  // 2) Q projection (256 rows) and fused K+V projection (32768 rows, ini_k read once)
  proj_kernel<false, 0><<<(BB * QQ) / 16, 256, 0, stream>>>(
      ini_q, Wtq, bq, Qh, nullptr, nullptr, nullptr);
  proj_kernel<true, 1><<<(BB * SS * TT) / 16, 256, 0, stream>>>(
      ini_k, Wtk, bk, Kh, Wtv, bv, Vt);
  // 3) attention: one block per (b,h,s)
  attn_kernel<<<BB * NHEAD * SS, 128, 0, stream>>>(Qh, Kh, Vt, mask, Res);
  // 4) output projection + LayerNorm
  oproj_ln_kernel<<<(BB * QQ * SS) / 16, 256, 0, stream>>>(
      Res, Wto, bo, gamma, beta, (float*)d_out);
}
